// MultiAttention_377957122666
// MI455X (gfx1250) — matrix operand order repaired and checked by reading, not compiled, not run on hardware
//
#include <hip/hip_runtime.h>

// ---------------------------------------------------------------------------
// Dual-branch cross attention, restructured (matmul associativity kills the
// N^3 cross-products):
//   out_r  = attn_r @ v_r ,  out_d = attn_d @ v_d
//   out_rd = a*u_r⊙(attn_r^T @ [(u_r u_d)⊙(attn_d @ (u_d⊙v_d))]) + (1-a)(attn_r@v_d + attn_d@v_d)
//   out_dr = symmetric
// GEMMs: v_wmma_f32_16x16x32_f16 (wave32). Hot attn streams: double-buffered
// GLOBAL_LOAD_ASYNC_TO_LDS_B128/B64 + s_wait_asynccnt (CDNA5 async DMA).
// ---------------------------------------------------------------------------

typedef __attribute__((ext_vector_type(16))) _Float16 v16h;
typedef __attribute__((ext_vector_type(8)))  _Float16 v8h;
typedef __attribute__((ext_vector_type(8)))  float    v8f;
typedef _Float16 f16;

#define ALPHA_F 0.6f
#define ONE_MINUS_ALPHA_F 0.4f
#define SCALE_F 0.17677669529663687f   // 32^-0.5

// ---- WMMA fragment helpers (ISA 7.12.2 layouts, wave32) --------------------
__device__ __forceinline__ v16h cat8(v8h lo, v8h hi) {
  return __builtin_shufflevector(lo, hi, 0,1,2,3,4,5,6,7,8,9,10,11,12,13,14,15);
}
// A fragment: 16x32 f16, lane m = lane&15, half = lane>>4.
__device__ __forceinline__ v16h frag_a(const f16* tile, int lda, int lane) {
  int m = lane & 15, h = lane >> 4;
  v8h lo = *(const v8h*)(tile + m * lda + 8 * h);
  v8h hi = *(const v8h*)(tile + m * lda + 16 + 8 * h);
  return cat8(lo, hi);
}
// B fragment: 32x16 f16 from LDS tile stored bt[n][k] (= B[k][n]).
__device__ __forceinline__ v16h frag_b(const f16* bt, int ldb, int lane) {
  int n = lane & 15, h = lane >> 4;
  v8h lo = *(const v8h*)(bt + n * ldb + 8 * h);
  v8h hi = *(const v8h*)(bt + n * ldb + 16 + 8 * h);
  return cat8(lo, hi);
}
__device__ __forceinline__ v8f wmma_f16(v16h a, v16h b, v8f c) {
  return __builtin_amdgcn_wmma_f32_16x16x32_f16(false, a, false, b, (short)0, c,
                                                false, false);
}

// ---- CDNA5 async global->LDS DMA (ASYNCcnt tracked) ------------------------
__device__ __forceinline__ void async_b128(const f16* g, f16* l) {
  unsigned lds = (unsigned)(unsigned long long)l;     // low 32 bits = LDS byte addr
  unsigned long long ga = (unsigned long long)g;
  asm volatile("global_load_async_to_lds_b128 %0, %1, off"
               :: "v"(lds), "v"(ga) : "memory");
}
__device__ __forceinline__ void async_b64(const f16* g, f16* l) {
  unsigned lds = (unsigned)(unsigned long long)l;
  unsigned long long ga = (unsigned long long)g;
  asm volatile("global_load_async_to_lds_b64 %0, %1, off"
               :: "v"(lds), "v"(ga) : "memory");
}
template <int N>
__device__ __forceinline__ void wait_async() {
  asm volatile("s_wait_asynccnt %0" :: "i"(N) : "memory");
}

// ---------------------------------------------------------------------------
// Kernel 1: Wt[k][n] = (f16) W[n][k]   (W: 768x256 fp32)
// ---------------------------------------------------------------------------
__global__ __launch_bounds__(256) void k_wt(const float* __restrict__ w,
                                            f16* __restrict__ wt) {
  int i = blockIdx.x * 256 + threadIdx.x;
  if (i >= 768 * 256) return;
  int n = i >> 8, k = i & 255;
  wt[k * 768 + n] = (f16)w[i];
}

// ---------------------------------------------------------------------------
// Kernel 2: QKV GEMM.  A = x[:, colOff:colOff+256] (4096x256 fp32, ld 512)
//           B = Wt (256x768 f16).
// Scatter -> Q[bh][n][d], K[bh][n][d] (row-major!), V[bh][n][d], Vt[bh][d][n]
// Block tile 64x128, 8 waves each 32x32 (2x2 WMMA tiles).
// ---------------------------------------------------------------------------
__global__ __launch_bounds__(256) void k_qkv(const float* __restrict__ x, int colOff,
                                             const f16* __restrict__ wt,
                                             f16* __restrict__ q,
                                             f16* __restrict__ kmat,
                                             f16* __restrict__ v,
                                             f16* __restrict__ vt) {
  __shared__ __align__(16) f16 sa[64][40];
  __shared__ __align__(16) f16 sb[128][40];   // transposed: sb[j][k]
  const int tid = threadIdx.x, lane = tid & 31, wid = tid >> 5;
  const int row0 = blockIdx.y * 64;
  const int col0 = blockIdx.x * 128;
  const int wr = wid >> 2, wc = wid & 3;

  v8f acc[2][2] = {};
  for (int kk = 0; kk < 256; kk += 32) {
    __syncthreads();
    {   // A: 64x32 fp32 -> f16
      int i = tid >> 2, k0 = (tid & 3) * 8;
      const float* src = x + (size_t)(row0 + i) * 512 + colOff + kk + k0;
      #pragma unroll
      for (int k = 0; k < 8; ++k) sa[i][k0 + k] = (f16)src[k];
    }
    {   // B: 32x128 f16 -> transposed LDS
      int k = tid >> 3, j0 = (tid & 7) * 16;
      const f16* src = wt + (size_t)(kk + k) * 768 + col0 + j0;
      #pragma unroll
      for (int j = 0; j < 16; ++j) sb[j0 + j][k] = src[j];
    }
    __syncthreads();
    v16h a0 = frag_a(&sa[wr * 32][0], 40, lane);
    v16h a1 = frag_a(&sa[wr * 32 + 16][0], 40, lane);
    v16h b0 = frag_b(&sb[wc * 32][0], 40, lane);
    v16h b1 = frag_b(&sb[wc * 32 + 16][0], 40, lane);
    acc[0][0] = wmma_f16(a0, b0, acc[0][0]);
    acc[0][1] = wmma_f16(a0, b1, acc[0][1]);
    acc[1][0] = wmma_f16(a1, b0, acc[1][0]);
    acc[1][1] = wmma_f16(a1, b1, acc[1][1]);
  }
  const int half = lane >> 4, nl = lane & 15;
  #pragma unroll
  for (int sr = 0; sr < 2; ++sr)
    #pragma unroll
    for (int sc = 0; sc < 2; ++sc) {
      int Cg = col0 + wc * 32 + sc * 16 + nl;
      int which = Cg >> 8, rem = Cg & 255, hh = rem >> 5, d = rem & 31;
      #pragma unroll
      for (int r = 0; r < 8; ++r) {
        int Rg = row0 + wr * 32 + sr * 16 + r + 8 * half;
        int b = Rg >> 10, n = Rg & 1023;
        size_t bh = (size_t)(b * 8 + hh);
        f16 val = (f16)acc[sr][sc][r];
        if (which == 0)      q[bh * 32768 + n * 32 + d] = val;
        else if (which == 1) kmat[bh * 32768 + n * 32 + d] = val;
        else {               v[bh * 32768 + n * 32 + d] = val;
                             vt[bh * 32768 + d * 1024 + n] = val; }
      }
    }
}

// ---------------------------------------------------------------------------
// Kernel 3: scores + softmax. One block = (bh, 16-row block), all 1024 cols.
// B operand of Q@K^T is row-major K -> straight 2D copy: async-DMA, 8 chunks
// of 128 cols, double-buffered. Then full-row softmax; writes attn, attn^T, u.
// ---------------------------------------------------------------------------
__global__ __launch_bounds__(256) void k_scores(const f16* __restrict__ q,
                                                const f16* __restrict__ kmat,
                                                f16* __restrict__ att,
                                                f16* __restrict__ attT,
                                                float* __restrict__ u) {
  __shared__ __align__(16) f16 sq[16][40];
  __shared__ __align__(16) f16 sbt[2][128][40];
  __shared__ __align__(16) f16 ss[16][1032];
  __shared__ float red[16][16];
  __shared__ float rowv[16];

  const int tid = threadIdx.x, lane = tid & 31, wid = tid >> 5;
  const int bh = blockIdx.x >> 6;
  const int row0 = (blockIdx.x & 63) * 16;
  const f16* qb = q + (size_t)bh * 32768;
  const f16* kb = kmat + (size_t)bh * 32768;

  for (int e = tid; e < 512; e += 256) {
    int i = e >> 5, k = e & 31;
    sq[i][k] = qb[(row0 + i) * 32 + k];
  }
  // async issue of a 128x32 K chunk (8KB): 2 x b128 per thread
  auto issueK = [&](int buf, int m0) {
    #pragma unroll
    for (int qd = 0; qd < 2; ++qd) {
      int e = tid + 256 * qd;
      int j = e >> 2, c = e & 3;
      async_b128(kb + (size_t)(m0 + j) * 32 + c * 8, &sbt[buf][j][c * 8]);
    }
  };
  issueK(0, 0);
  for (int cc = 0; cc < 8; ++cc) {
    __syncthreads();                                   // (A)
    if (cc + 1 < 8) issueK((cc + 1) & 1, (cc + 1) * 128);
    if (cc + 1 < 8) wait_async<2>(); else wait_async<0>();
    __syncthreads();                                   // (B)
    v16h a = frag_a(&sq[0][0], 40, lane);
    int jb = wid * 16;                                 // one 16-col tile per wave
    v16h b = frag_b(&sbt[cc & 1][jb][0], 40, lane);
    v8f d = {};
    d = wmma_f16(a, b, d);
    int half = lane >> 4, nl = lane & 15;
    #pragma unroll
    for (int r = 0; r < 8; ++r)
      ss[r + 8 * half][cc * 128 + jb + nl] = (f16)(d[r] * SCALE_F);
  }
  __syncthreads();

  // row softmax: 16 threads per row
  const int row = tid >> 4, ln = tid & 15;
  float mx = -3.0e38f;
  for (int i = 0; i < 64; ++i) mx = fmaxf(mx, (float)ss[row][ln + 16 * i]);
  red[row][ln] = mx; __syncthreads();
  if (ln == 0) { float m = red[row][0];
                 for (int i = 1; i < 16; ++i) m = fmaxf(m, red[row][i]);
                 rowv[row] = m; }
  __syncthreads();
  const float M = rowv[row];
  float sum = 0.f;
  for (int i = 0; i < 64; ++i) sum += __expf((float)ss[row][ln + 16 * i] - M);
  __syncthreads();
  red[row][ln] = sum; __syncthreads();
  if (ln == 0) { float s = 0.f;
                 for (int i = 0; i < 16; ++i) s += red[row][i];
                 rowv[row] = s; }
  __syncthreads();
  const float inv = 1.0f / rowv[row];
  const int nrow = row0 + row;
  f16* arow = att  + (size_t)bh * 1048576 + (size_t)nrow * 1024;
  f16* atb  = attT + (size_t)bh * 1048576;
  float psum = 0.f;
  for (int i = 0; i < 64; ++i) {
    int c = ln + 16 * i;
    float p = __expf((float)ss[row][c] - M) * inv;
    psum += p;
    f16 ph = (f16)p;
    arow[c] = ph;
    atb[(size_t)c * 1024 + nrow] = ph;
  }
  __syncthreads();
  red[row][ln] = psum; __syncthreads();
  if (ln == 0) { float s = 0.f;
                 for (int i = 0; i < 16; ++i) s += red[row][i];
                 u[bh * 1024 + nrow] = rsqrtf(s); }
}

// ---------------------------------------------------------------------------
// Kernel 4 (pass 1): attn @ [v_same | v_other | u_same⊙v_same]  (1024x96, K=1024)
//   g0 -> out_same (d_out, f32); g1 -> av_cross (ws, f32)
//   g2 -> X^T (ws, f16) = transpose of (u_r·u_d)⊙(attn @ (u_same⊙v_same))
// A tiles (the 67MB attn stream): async DMA, double-buffered.
// ---------------------------------------------------------------------------
__global__ __launch_bounds__(256) void k_pass1(const f16* __restrict__ att,
                                               const f16* __restrict__ v_same,
                                               const f16* __restrict__ v_oth,
                                               const float* __restrict__ u_same,
                                               const float* __restrict__ u_r,
                                               const float* __restrict__ u_d,
                                               float* __restrict__ out_same,
                                               float* __restrict__ av_cross,
                                               f16* __restrict__ xtout) {
  __shared__ __align__(16) f16 sa[2][128][40];
  __shared__ __align__(16) f16 sbt[96][40];
  const int tid = threadIdx.x, lane = tid & 31, wid = tid >> 5;
  const int bh = blockIdx.x >> 3;
  const int row0 = (blockIdx.x & 7) * 128;
  const f16* ab = att + (size_t)bh * 1048576;
  const f16* vs = v_same + (size_t)bh * 32768;
  const f16* vo = v_oth  + (size_t)bh * 32768;
  const float* us = u_same + bh * 1024;

  auto issueA = [&](int buf, int kk) {                 // 128x32 tile, 2 x b128/thread
    #pragma unroll
    for (int qd = 0; qd < 2; ++qd) {
      int e = tid + 256 * qd;
      int i = e >> 2, c = e & 3;
      async_b128(ab + (size_t)(row0 + i) * 1024 + kk + c * 8, &sa[buf][i][c * 8]);
    }
  };

  v8f acc[6] = {};
  issueA(0, 0);
  for (int s = 0; s < 32; ++s) {
    const int kk = s * 32;
    __syncthreads();                                   // (A)
    if (s + 1 < 32) issueA((s + 1) & 1, kk + 32);
    // B 32x96 -> sbt[j][k] (sync; g2 scaled by u_same[k])
    for (int e = tid; e < 3072; e += 256) {
      int k = e & 31, j = e >> 5;
      int g = j >> 5, d = j & 31;
      float val;
      if (g == 0)      val = (float)vs[(kk + k) * 32 + d];
      else if (g == 1) val = (float)vo[(kk + k) * 32 + d];
      else             val = (float)vs[(kk + k) * 32 + d] * us[kk + k];
      sbt[j][k] = (f16)val;
    }
    if (s + 1 < 32) wait_async<2>(); else wait_async<0>();
    __syncthreads();                                   // (B)
    v16h a = frag_a(&sa[s & 1][wid * 16][0], 40, lane);
    #pragma unroll
    for (int t = 0; t < 6; ++t) {
      v16h b = frag_b(&sbt[t * 16][0], 40, lane);
      acc[t] = wmma_f16(a, b, acc[t]);
    }
  }
  const int half = lane >> 4, nl = lane & 15;
  const int b = bh >> 3, hh = bh & 7;
  #pragma unroll
  for (int t = 0; t < 6; ++t) {
    int j = t * 16 + nl, g = j >> 5, d = j & 31;
    #pragma unroll
    for (int r = 0; r < 8; ++r) {
      int n = row0 + wid * 16 + r + 8 * half;
      float val = acc[t][r];
      if (g == 0) {
        out_same[((size_t)b * 1024 + n) * 256 + hh * 32 + d] = val;
      } else if (g == 1) {
        av_cross[(size_t)bh * 32768 + n * 32 + d] = val;
      } else {
        float sc = u_r[bh * 1024 + n] * u_d[bh * 1024 + n];
        xtout[(size_t)bh * 32768 + (size_t)d * 1024 + n] = (f16)(val * sc);
      }
    }
  }
}

// ---------------------------------------------------------------------------
// Kernel 5 (pass 2 + combine): Z = attn^T @ X (1024x32, K=1024)
//   out = ALPHA * u_out[n] * Z + (1-ALPHA) * (av_a + av_b)
// Both A (attn^T) and B (X^T, contiguous) tiles async-DMA, double-buffered.
// ---------------------------------------------------------------------------
__global__ __launch_bounds__(256) void k_pass2(const f16* __restrict__ attT,
                                               const f16* __restrict__ xt,
                                               const float* __restrict__ u_out,
                                               const float* __restrict__ av_a,
                                               const float* __restrict__ av_b,
                                               float* __restrict__ outp) {
  __shared__ __align__(16) f16 sa[2][128][40];
  __shared__ __align__(16) f16 sbt[2][32][40];
  const int tid = threadIdx.x, lane = tid & 31, wid = tid >> 5;
  const int bh = blockIdx.x >> 3;
  const int row0 = (blockIdx.x & 7) * 128;
  const f16* ab = attT + (size_t)bh * 1048576;
  const f16* xb = xt + (size_t)bh * 32768;

  auto issueA = [&](int buf, int kk) {
    #pragma unroll
    for (int qd = 0; qd < 2; ++qd) {
      int e = tid + 256 * qd;
      int i = e >> 2, c = e & 3;
      async_b128(ab + (size_t)(row0 + i) * 1024 + kk + c * 8, &sa[buf][i][c * 8]);
    }
  };
  auto issueB = [&](int buf, int kk) {                 // 32x32 tile, 1 x b64/thread
    int j = tid >> 3, c = tid & 7;
    async_b64(xb + (size_t)j * 1024 + kk + c * 4, &sbt[buf][j][c * 4]);
  };

  v8f acc[2] = {};
  issueA(0, 0);
  issueB(0, 0);
  for (int s = 0; s < 32; ++s) {
    __syncthreads();                                   // (A)
    if (s + 1 < 32) { issueA((s + 1) & 1, (s + 1) * 32);
                      issueB((s + 1) & 1, (s + 1) * 32); }
    if (s + 1 < 32) wait_async<3>(); else wait_async<0>();
    __syncthreads();                                   // (B)
    v16h a = frag_a(&sa[s & 1][wid * 16][0], 40, lane);
    #pragma unroll
    for (int t = 0; t < 2; ++t) {
      v16h b = frag_b(&sbt[s & 1][t * 16][0], 40, lane);
      acc[t] = wmma_f16(a, b, acc[t]);
    }
  }
  const int half = lane >> 4, nl = lane & 15;
  const int b = bh >> 3, hh = bh & 7;
  #pragma unroll
  for (int t = 0; t < 2; ++t) {
    int d = t * 16 + nl;
    #pragma unroll
    for (int r = 0; r < 8; ++r) {
      int n = row0 + wid * 16 + r + 8 * half;
      size_t oidx = ((size_t)b * 1024 + n) * 256 + hh * 32 + d;
      float z = ALPHA_F * u_out[bh * 1024 + n] * acc[t][r]
              + ONE_MINUS_ALPHA_F * (av_a[(size_t)bh * 32768 + n * 32 + d] + av_b[oidx]);
      outp[oidx] = z;
    }
  }
}

// ---------------------------------------------------------------------------
extern "C" void kernel_launch(void* const* d_in, const int* in_sizes, int n_in,
                              void* d_out, int out_size, void* d_ws, size_t ws_size,
                              hipStream_t stream) {
  (void)in_sizes; (void)n_in; (void)out_size; (void)ws_size;
  const float* x  = (const float*)d_in[0];
  const float* wr = (const float*)d_in[1];
  const float* wd = (const float*)d_in[2];

  char* ws = (char*)d_ws;
  size_t off = 0;
  auto alloc = [&](size_t bytes) -> char* {
    char* p = ws + off;
    off += (bytes + 255) & ~(size_t)255;
    return p;
  };
  const size_t SZ_WT  = (size_t)256 * 768 * 2;     // f16
  const size_t SZ_QV  = (size_t)32 * 32768 * 2;    // f16 [bh][1024][32]
  const size_t SZ_ATT = (size_t)32 * 1048576 * 2;  // f16 [bh][1024][1024]
  const size_t SZ_U   = (size_t)32 * 1024 * 4;     // f32
  const size_t SZ_AV  = (size_t)32 * 32768 * 4;    // f32

  f16* wt_r   = (f16*)alloc(SZ_WT);
  f16* wt_d   = (f16*)alloc(SZ_WT);
  f16* q_r    = (f16*)alloc(SZ_QV);
  f16* k_r    = (f16*)alloc(SZ_QV);
  f16* v_r    = (f16*)alloc(SZ_QV);
  f16* vt_r   = (f16*)alloc(SZ_QV);
  f16* q_d    = (f16*)alloc(SZ_QV);
  f16* k_d    = (f16*)alloc(SZ_QV);
  f16* v_d    = (f16*)alloc(SZ_QV);
  f16* vt_d   = (f16*)alloc(SZ_QV);
  f16* att_r  = (f16*)alloc(SZ_ATT);
  f16* attT_r = (f16*)alloc(SZ_ATT);
  f16* att_d  = (f16*)alloc(SZ_ATT);
  f16* attT_d = (f16*)alloc(SZ_ATT);
  float* u_r  = (float*)alloc(SZ_U);
  float* u_d  = (float*)alloc(SZ_U);
  float* av_rd = (float*)alloc(SZ_AV);   // attn_r @ v_d
  float* av_dr = (float*)alloc(SZ_AV);   // attn_d @ v_r
  f16* xt_rd  = (f16*)alloc(SZ_QV);      // transpose of (u_r u_d)⊙(attn_d @ (u_d⊙v_d))
  f16* xt_dr  = (f16*)alloc(SZ_QV);      // transpose of (u_r u_d)⊙(attn_r @ (u_r⊙v_r))

  float* out_dr = (float*)d_out;
  float* out_rd = out_dr + (size_t)1048576;
  float* out_r  = out_dr + (size_t)2 * 1048576;
  float* out_d  = out_dr + (size_t)3 * 1048576;

  dim3 blk(256);
  k_wt<<<768, blk, 0, stream>>>(wr, wt_r);
  k_wt<<<768, blk, 0, stream>>>(wd, wt_d);

  dim3 gq(6, 64);   // 768/128 cols x 4096/64 rows
  k_qkv<<<gq, blk, 0, stream>>>(x, 0,   wt_r, q_r, k_r, v_r, vt_r);
  k_qkv<<<gq, blk, 0, stream>>>(x, 256, wt_d, q_d, k_d, v_d, vt_d);

  k_scores<<<2048, blk, 0, stream>>>(q_r, k_r, att_r, attT_r, u_r);
  k_scores<<<2048, blk, 0, stream>>>(q_d, k_d, att_d, attT_d, u_d);

  // pass1 r: attn_r @ [v_r | v_d | u_r⊙v_r] -> out_r, av_rd, xt_dr
  k_pass1<<<256, blk, 0, stream>>>(att_r, v_r, v_d, u_r, u_r, u_d, out_r, av_rd, xt_dr);
  // pass1 d: attn_d @ [v_d | v_r | u_d⊙v_d] -> out_d, av_dr, xt_rd
  k_pass1<<<256, blk, 0, stream>>>(att_d, v_d, v_r, u_d, u_r, u_d, out_d, av_dr, xt_rd);

  // out_rd = a*u_r⊙(attn_r^T @ X_rd) + (1-a)*(av_rd + out_d)
  k_pass2<<<256, blk, 0, stream>>>(attT_r, xt_rd, u_r, av_rd, out_d, out_rd);
  // out_dr = a*u_d⊙(attn_d^T @ X_dr) + (1-a)*(av_dr + out_r)
  k_pass2<<<256, blk, 0, stream>>>(attT_d, xt_dr, u_d, av_dr, out_r, out_dr);
}
